// MPNDiffEncoder_38173669326911
// MI455X (gfx1250) — matrix operand order repaired and checked by reading, not compiled, not run on hardware
//
#include <hip/hip_runtime.h>
#include <hip/hip_bf16.h>

typedef unsigned short u16;
typedef unsigned int u32;

typedef __attribute__((ext_vector_type(16))) __bf16 v16bf;
typedef __attribute__((ext_vector_type(8)))  __bf16 v8bf;
typedef __attribute__((ext_vector_type(8)))  float  v8f;

#define N_ATOMS   100000
#define N_BONDS   300000
#define ATOM_FDIM 133
#define BOND_FDIM 14
#define FB_DIM    147      // ATOM_FDIM + BOND_FDIM
#define HIDDEN    512
#define MAX_NB    6
#define N_MOLS    1000

// padded K dims (multiples of 32)
#define K_WI      160      // 133 -> 160
#define K_WH      544      // 526 -> 544
#define K_WO      672      // 645 -> 672

// N-tiles per wave (register blocking): one A fragment feeds NB WMMAs.
#define NB        8
#define NGROUPS   (32 / NB)   // 4 N-tile groups covering 512 columns

__device__ __forceinline__ u16 f2bf(float f) {
    u32 u = __float_as_uint(f);
    u32 r = (u + 0x7FFFu + ((u >> 16) & 1u)) >> 16;   // RNE
    return (u16)r;
}
__device__ __forceinline__ float bf2f(u32 h) {
    return __uint_as_float(h << 16);
}

// ---------------------------------------------------------------------------
// Pack a [Kreal,512] f32 weight matrix into per-lane WMMA B fragments (bf16),
// zero-padded to k32*32 in K. Layout: [kb][ntile][lane][16 K-values].
// lane 0..15  -> N = nt*16+lane,    K = kb*32 + {0..7, 16..23}
// lane 16..31 -> N = nt*16+lane-16, K = kb*32 + {8..15, 24..31}
// ---------------------------------------------------------------------------
__global__ void pack_w_kernel(const float* __restrict__ W, u16* __restrict__ dst,
                              int Kreal, int k32) {
    int tid = blockIdx.x * 256 + threadIdx.x;
    int total = k32 * 1024;                 // k32 * 32 ntiles * 32 lanes
    if (tid >= total) return;
    int kb   = tid >> 10;
    int nt   = (tid >> 5) & 31;
    int lane = tid & 31;
    size_t o = (size_t)tid * 16;
    int n = nt * 16 + (lane & 15);
    int kbase = kb * 32 + ((lane >> 4) << 3);
    #pragma unroll
    for (int j = 0; j < 16; ++j) {
        int k = kbase + j + ((j >> 3) << 3);
        float v = (k < Kreal) ? W[(size_t)k * HIDDEN + n] : 0.0f;
        dst[o + j] = f2bf(v);
    }
}

// atom_features f32 [N,133] -> bf16 [N,160] zero padded
__global__ void pack_af_kernel(const float* __restrict__ af, u16* __restrict__ dst) {
    long long tid = (long long)blockIdx.x * 256 + threadIdx.x;
    if (tid >= (long long)N_ATOMS * K_WI) return;
    int row = (int)(tid / K_WI);
    int c   = (int)(tid % K_WI);
    dst[tid] = (c < ATOM_FDIM) ? f2bf(af[(size_t)row * ATOM_FDIM + c]) : (u16)0;
}

// ---------------------------------------------------------------------------
// bf16 WMMA GEMM: C[M,512] = A[M,lda] x Bpacked. Each wave: one 16-row M strip
// x NB=8 consecutive 16-col N tiles (8 accumulators); A fragment loaded once
// per K step and reused for 8 WMMAs.
// mode 0: outRaw = acc (bf16),  outMsg = relu(acc) (bf16)
// mode 1: outMsg = relu(bf(inpBF) + acc) (bf16)
// mode 2: outMsg = relu(acc + bias[n]) (bf16)
// ---------------------------------------------------------------------------
__global__ __launch_bounds__(256)
void gemm_bf16_kernel(const u16* __restrict__ A, int lda,
                      const u16* __restrict__ Bp, int k32,
                      u16* __restrict__ outMsg,
                      u16* __restrict__ outRaw,
                      const u16* __restrict__ inpBF,
                      const float* __restrict__ bias,
                      int mode, int mtiles) {
    int wave = threadIdx.x >> 5;
    int lane = threadIdx.x & 31;
    int tile = blockIdx.x * 8 + wave;
    if (tile >= mtiles * NGROUPS) return;
    int tm  = tile >> 2;                 // M tile (16 rows)
    int tng = tile & (NGROUPS - 1);      // group of NB consecutive N tiles
    int half = lane >> 4;
    int r16  = lane & 15;

    const u16* arow  = A + (size_t)(tm * 16 + r16) * lda;
    // lane base within packed B for N tile (tng*NB + q):
    const u16* bbase = Bp + ((size_t)(tng * NB) * 32 + lane) * 16;

    v8f c[NB];
    #pragma unroll
    for (int q = 0; q < NB; ++q) c[q] = (v8f){};

    for (int kb = 0; kb < k32; ++kb) {
        union { v16bf v; v8bf h[2]; } a;
        int k0 = kb * 32 + half * 8;
        a.h[0] = *reinterpret_cast<const v8bf*>(arow + k0);
        a.h[1] = *reinterpret_cast<const v8bf*>(arow + k0 + 16);
        const u16* bkb = bbase + (size_t)kb * (32 * 32 * 16);
        #pragma unroll
        for (int q = 0; q < NB; ++q) {
            v16bf b = *reinterpret_cast<const v16bf*>(bkb + (size_t)q * (32 * 16));
            c[q] = __builtin_amdgcn_wmma_f32_16x16x32_bf16(false, a.v, false, b,
                                                           (short)0, c[q], false, false);
        }
    }

    int row0 = tm * 16 + half * 8;
    #pragma unroll
    for (int q = 0; q < NB; ++q) {
        int col = (tng * NB + q) * 16 + r16;
        #pragma unroll
        for (int r = 0; r < 8; ++r) {
            size_t idx = (size_t)(row0 + r) * HIDDEN + col;
            float v = c[q][r];
            if (mode == 0) {
                outRaw[idx] = f2bf(v);
                outMsg[idx] = f2bf(v > 0.0f ? v : 0.0f);
            } else if (mode == 1) {
                float s = bf2f(inpBF[idx]) + v;
                outMsg[idx] = f2bf(s > 0.0f ? s : 0.0f);
            } else {
                float s = v + bias[col];
                outMsg[idx] = f2bf(s > 0.0f ? s : 0.0f);
            }
        }
    }
}

// ---------------------------------------------------------------------------
// Per-depth gather: cat[i, 0:512]  = sum_j message[a2a[i,j]]       (bf16)
//                   cat[i, 512:526]= sum_j f_bonds[a2b[i,j], 133:] (bf16)
//                   cat[i, 526:544]= 0
// 64 threads per atom; thread t handles msg cols [8t, 8t+8).
// ---------------------------------------------------------------------------
__global__ void gather_iter_kernel(const u16* __restrict__ msg,
                                   const float* __restrict__ fb,
                                   const int* __restrict__ a2a,
                                   const int* __restrict__ a2b,
                                   u16* __restrict__ cat) {
    int i = blockIdx.x;
    int t = threadIdx.x;                       // 0..63
    const int* na = a2a + (size_t)i * MAX_NB;
    float acc[8] = {0,0,0,0,0,0,0,0};
    #pragma unroll
    for (int j = 0; j < MAX_NB; ++j) {
        const uint4 u = *reinterpret_cast<const uint4*>(
            msg + (size_t)na[j] * HIDDEN + t * 8);
        u32 w0 = u.x, w1 = u.y, w2 = u.z, w3 = u.w;
        acc[0] += bf2f(w0 & 0xffffu); acc[1] += bf2f(w0 >> 16);
        acc[2] += bf2f(w1 & 0xffffu); acc[3] += bf2f(w1 >> 16);
        acc[4] += bf2f(w2 & 0xffffu); acc[5] += bf2f(w2 >> 16);
        acc[6] += bf2f(w3 & 0xffffu); acc[7] += bf2f(w3 >> 16);
    }
    uint4 o;
    o.x = (u32)f2bf(acc[0]) | ((u32)f2bf(acc[1]) << 16);
    o.y = (u32)f2bf(acc[2]) | ((u32)f2bf(acc[3]) << 16);
    o.z = (u32)f2bf(acc[4]) | ((u32)f2bf(acc[5]) << 16);
    o.w = (u32)f2bf(acc[6]) | ((u32)f2bf(acc[7]) << 16);
    *reinterpret_cast<uint4*>(cat + (size_t)i * K_WH + t * 8) = o;

    if (t < BOND_FDIM) {
        const int* nb = a2b + (size_t)i * MAX_NB;
        float s = 0.0f;
        #pragma unroll
        for (int j = 0; j < MAX_NB; ++j)
            s += fb[(size_t)nb[j] * FB_DIM + ATOM_FDIM + t];
        cat[(size_t)i * K_WH + HIDDEN + t] = f2bf(s);
    } else if (t < 32) {
        cat[(size_t)i * K_WH + HIDDEN + t] = 0;   // cols 526..543 pad
    }
}

// Final gather: A3[i] = [bf16(atom_features[i]) | bf16(sum_j msg[a2a[i,j]]) | 0pad]
__global__ void gather_final_kernel(const u16* __restrict__ msg,
                                    const float* __restrict__ af,
                                    const int* __restrict__ a2a,
                                    u16* __restrict__ A3) {
    int i = blockIdx.x;
    int t = threadIdx.x;                       // 0..671
    u16 v;
    if (t < ATOM_FDIM) {
        v = f2bf(af[(size_t)i * ATOM_FDIM + t]);
    } else if (t < ATOM_FDIM + HIDDEN) {
        int c = t - ATOM_FDIM;
        const int* na = a2a + (size_t)i * MAX_NB;
        float s = 0.0f;
        #pragma unroll
        for (int j = 0; j < MAX_NB; ++j)
            s += bf2f(msg[(size_t)na[j] * HIDDEN + c]);
        v = f2bf(s);
    } else {
        v = 0;
    }
    A3[(size_t)i * K_WO + t] = v;
}

__global__ void zero_f_kernel(float* __restrict__ p, int n) {
    int i = blockIdx.x * 256 + threadIdx.x;
    if (i < n) p[i] = 0.0f;
}

// One wave per atom: molecule segment accumulation with f32 atomics.
__global__ void seg_acc_kernel(const u16* __restrict__ hid,
                               const float* __restrict__ af,
                               const int* __restrict__ mol,
                               float* __restrict__ sums,
                               float* __restrict__ cnt,
                               float* __restrict__ nz) {
    int i = blockIdx.x;
    int lane = threadIdx.x;                    // 0..31
    int m = mol[i];
    float s = 0.0f;
    for (int c = lane; c < ATOM_FDIM; c += 32)
        s += af[(size_t)i * ATOM_FDIM + c];
    for (int off = 16; off > 0; off >>= 1)
        s += __shfl_down(s, off);
    if (lane == 0) {
        atomicAdd(&cnt[m], 1.0f);
        if (s > 0.0f) atomicAdd(&nz[m], 1.0f);
    }
    for (int h = lane; h < HIDDEN; h += 32)
        atomicAdd(&sums[(size_t)m * HIDDEN + h],
                  bf2f(hid[(size_t)i * HIDDEN + h]));
}

__global__ void finalize_kernel(const float* __restrict__ sums,
                                const float* __restrict__ cnt,
                                const float* __restrict__ nz,
                                float* __restrict__ out) {
    int m = blockIdx.x;
    int t = threadIdx.x;                       // 0..511
    float c = cnt[m];
    float d = c > 1.0f ? c : 1.0f;
    out[(size_t)m * HIDDEN + t] = sums[(size_t)m * HIDDEN + t] / d;
    if (t == 0) {
        out[(size_t)N_MOLS * HIDDEN + m * 2 + 0] = nz[m];
        out[(size_t)N_MOLS * HIDDEN + m * 2 + 1] = c;
    }
}

extern "C" void kernel_launch(void* const* d_in, const int* in_sizes, int n_in,
                              void* d_out, int out_size, void* d_ws, size_t ws_size,
                              hipStream_t stream) {
    const float* af  = (const float*)d_in[0];
    const float* fb  = (const float*)d_in[1];
    const int*   a2a = (const int*)d_in[2];
    const int*   a2b = (const int*)d_in[3];
    const int*   mol = (const int*)d_in[4];
    const float* Wi  = (const float*)d_in[5];
    const float* Wh  = (const float*)d_in[6];
    const float* Wo  = (const float*)d_in[7];
    const float* bo  = (const float*)d_in[8];
    float* out = (float*)d_out;

    char* ws = (char*)d_ws;
    size_t off = 0;
    auto alloc = [&](size_t bytes) -> void* {
        void* p = ws + off;
        off += (bytes + 255) & ~(size_t)255;
        return p;
    };
    u16*   inp  = (u16*)alloc((size_t)N_ATOMS * HIDDEN * 2); // pre-relu W_i output (bf16)
    u16*   msg  = (u16*)alloc((size_t)N_ATOMS * HIDDEN * 2); // message / final hiddens (bf16)
    u16*   cat  = (u16*)alloc((size_t)N_ATOMS * K_WO * 2);   // gathered A (bf16, max 672 wide)
    u16*   afb  = (u16*)alloc((size_t)N_ATOMS * K_WI * 2);   // atom features bf16 padded
    u16*   wiP  = (u16*)alloc((size_t)(K_WI / 32) * 1024 * 16 * 2);
    u16*   whP  = (u16*)alloc((size_t)(K_WH / 32) * 1024 * 16 * 2);
    u16*   woP  = (u16*)alloc((size_t)(K_WO / 32) * 1024 * 16 * 2);
    float* sums = (float*)alloc((size_t)N_MOLS * HIDDEN * 4);
    float* cnt  = (float*)alloc((size_t)N_MOLS * 4);
    float* nz   = (float*)alloc((size_t)N_MOLS * 4);

    const int MT = N_ATOMS / 16;                      // 6250 M tiles
    const int GEMM_BLOCKS = (MT * NGROUPS + 7) / 8;   // 8 waves per block

    // weight / feature packing
    pack_w_kernel<<<((K_WI / 32) * 1024 + 255) / 256, 256, 0, stream>>>(Wi, wiP, ATOM_FDIM, K_WI / 32);
    pack_w_kernel<<<((K_WH / 32) * 1024 + 255) / 256, 256, 0, stream>>>(Wh, whP, HIDDEN + BOND_FDIM, K_WH / 32);
    pack_w_kernel<<<((K_WO / 32) * 1024 + 255) / 256, 256, 0, stream>>>(Wo, woP, ATOM_FDIM + HIDDEN, K_WO / 32);
    pack_af_kernel<<<(int)(((size_t)N_ATOMS * K_WI + 255) / 256), 256, 0, stream>>>(af, afb);

    // inp = af @ W_i ; message = relu(inp)
    gemm_bf16_kernel<<<GEMM_BLOCKS, 256, 0, stream>>>(afb, K_WI, wiP, K_WI / 32,
                                                      msg, inp, nullptr, nullptr, 0, MT);

    // DEPTH-1 = 3 message passing steps
    for (int it = 0; it < 3; ++it) {
        gather_iter_kernel<<<N_ATOMS, 64, 0, stream>>>(msg, fb, a2a, a2b, cat);
        gemm_bf16_kernel<<<GEMM_BLOCKS, 256, 0, stream>>>(cat, K_WH, whP, K_WH / 32,
                                                          msg, nullptr, inp, nullptr, 1, MT);
    }

    // readout
    gather_final_kernel<<<N_ATOMS, K_WO, 0, stream>>>(msg, af, a2a, cat);
    gemm_bf16_kernel<<<GEMM_BLOCKS, 256, 0, stream>>>(cat, K_WO, woP, K_WO / 32,
                                                      msg, nullptr, nullptr, bo, 2, MT);

    // per-molecule mean + counts
    zero_f_kernel<<<(N_MOLS * HIDDEN + 255) / 256, 256, 0, stream>>>(sums, N_MOLS * HIDDEN);
    zero_f_kernel<<<(N_MOLS + 255) / 256, 256, 0, stream>>>(cnt, N_MOLS);
    zero_f_kernel<<<(N_MOLS + 255) / 256, 256, 0, stream>>>(nz, N_MOLS);
    seg_acc_kernel<<<N_ATOMS, 32, 0, stream>>>(msg, af, mol, sums, cnt, nz);
    finalize_kernel<<<N_MOLS, HIDDEN, 0, stream>>>(sums, cnt, nz, out);
    (void)in_sizes; (void)n_in; (void)out_size; (void)ws_size;
}
